// latentMambaDQNNetwork_8615704396428
// MI455X (gfx1250) — matrix operand order repaired and checked by reading, not compiled, not run on hardware
//
#include <hip/hip_runtime.h>
#include <hip/hip_bf16.h>
#include <cstdint>

// ---- problem constants (from reference) ----
#define BSZ     64
#define SLEN    512
#define DIN     128
#define MLPH    256
#define DMODEL  128
#define DINNER  256      // 2*DMODEL
#define DSTATE  64
#define DCONV   4
#define DTRANK  8
#define NACT    18
#define NTOK    (BSZ*SLEN)          // 32768
#define XPROJN  (DTRANK + 2*DSTATE) // 136
#define XPROJP  160                 // padded to multiple of 32

typedef __attribute__((ext_vector_type(16))) __bf16 bf16x16;
typedef __attribute__((ext_vector_type(8)))  float  floatx8;

__device__ __forceinline__ float sigmoidf_(float x) { return 1.0f / (1.0f + __expf(-x)); }

// ---------------- f32 -> bf16 conversion ----------------
__global__ void cvt_f32_to_bf16(const float* __restrict__ in, __bf16* __restrict__ out, int n) {
    int i = blockIdx.x * blockDim.x + threadIdx.x;
    if (i < n) out[i] = (__bf16)in[i];
}

// f32[rows,cols] -> bf16[rowsPad,cols], zero-filled padding rows
__global__ void cvt_pad_bf16(const float* __restrict__ in, __bf16* __restrict__ out,
                             int rows, int cols, int rowsPad) {
    int i = blockIdx.x * blockDim.x + threadIdx.x;
    if (i >= rowsPad * cols) return;
    int r = i / cols;
    out[i] = (r < rows) ? (__bf16)in[i] : (__bf16)0.0f;
}

// ---------------- WMMA GEMM: C[M,N] = act(A[M,K] @ W[N,K]^T + bias) ----------------
// LDS-free, explicitly software-pipelined with NAMED stage registers (no
// dynamically indexed arrays -> everything stays in VGPRs). Each wave owns a
// 16x32 output tile: one shared A fragment + two B fragments -> two
// v_wmma_f32_16x16x32_bf16 per 32-k step; loop advances 64 k per iteration
// with stage-1 loads in flight while stage-0 WMMAs execute and vice versa.
// Requires: M%16==0, N%32==0 (pre-padded), K%64==0, 16B-aligned bases.
union Frag { bf16x16 v; uint4 q[2]; };

__global__ void wmma_gemm(const __bf16* __restrict__ A, const __bf16* __restrict__ W,
                          const float* __restrict__ bias,
                          float* __restrict__ Cf, __bf16* __restrict__ Cb,
                          int M, int N, int K, int Ntiles2, int act) {
    int tile = blockIdx.x * 4 + (threadIdx.x >> 5);
    if (tile >= (M >> 4) * Ntiles2) return;
    int tm = tile / Ntiles2, tn = tile % Ntiles2;
    int m0 = tm * 16, n0 = tn * 32;
    int lane = threadIdx.x & 31;            // 0..31 (wave32)

    // A fragment: lane = M row (mod 16); k-run of 16: 0-15 (lanes<16) / 16-31
    // B fragment: lane = N row (mod 16); k-run of 16: 0-15 (lanes<16) / 16-31
    const __bf16* arow  = A + (size_t)(m0 + (lane & 15)) * K;
    const __bf16* wrow0 = W + (size_t)(n0 + (lane & 15)) * K;
    const __bf16* wrow1 = wrow0 + (size_t)16 * K;
    int ka0 = (lane < 16) ? 0 : 16;
    int ka1 = (lane < 16) ? 8 : 24;
    int kb  = (lane < 16) ? 0 : 16;

    Frag a0, p0, r0, a1, p1, r1;            // two named pipeline stages
    floatx8 acc0 = {}, acc1 = {};

    // prologue: stage 0 at k=0
    a0.q[0] = *reinterpret_cast<const uint4*>(arow  + ka0);
    a0.q[1] = *reinterpret_cast<const uint4*>(arow  + ka1);
    p0.q[0] = *reinterpret_cast<const uint4*>(wrow0 + kb);
    p0.q[1] = *reinterpret_cast<const uint4*>(wrow0 + kb + 8);
    r0.q[0] = *reinterpret_cast<const uint4*>(wrow1 + kb);
    r0.q[1] = *reinterpret_cast<const uint4*>(wrow1 + kb + 8);

    for (int k0 = 0; k0 < K; k0 += 64) {
        // stage 1 loads (k0+32) go in flight over the stage-0 WMMAs
        a1.q[0] = *reinterpret_cast<const uint4*>(arow  + k0 + 32 + ka0);
        a1.q[1] = *reinterpret_cast<const uint4*>(arow  + k0 + 32 + ka1);
        p1.q[0] = *reinterpret_cast<const uint4*>(wrow0 + k0 + 32 + kb);
        p1.q[1] = *reinterpret_cast<const uint4*>(wrow0 + k0 + 32 + kb + 8);
        r1.q[0] = *reinterpret_cast<const uint4*>(wrow1 + k0 + 32 + kb);
        r1.q[1] = *reinterpret_cast<const uint4*>(wrow1 + k0 + 32 + kb + 8);

        acc0 = __builtin_amdgcn_wmma_f32_16x16x32_bf16(
            false, a0.v, false, p0.v, (short)0, acc0, false, false);
        acc1 = __builtin_amdgcn_wmma_f32_16x16x32_bf16(
            false, a0.v, false, r0.v, (short)0, acc1, false, false);

        if (k0 + 64 < K) {
            // refill stage 0 (k0+64) over the stage-1 WMMAs
            a0.q[0] = *reinterpret_cast<const uint4*>(arow  + k0 + 64 + ka0);
            a0.q[1] = *reinterpret_cast<const uint4*>(arow  + k0 + 64 + ka1);
            p0.q[0] = *reinterpret_cast<const uint4*>(wrow0 + k0 + 64 + kb);
            p0.q[1] = *reinterpret_cast<const uint4*>(wrow0 + k0 + 64 + kb + 8);
            r0.q[0] = *reinterpret_cast<const uint4*>(wrow1 + k0 + 64 + kb);
            r0.q[1] = *reinterpret_cast<const uint4*>(wrow1 + k0 + 64 + kb + 8);
            __builtin_prefetch(arow  + k0 + 128, 0, 1);
            __builtin_prefetch(wrow0 + k0 + 128, 0, 1);
            __builtin_prefetch(wrow1 + k0 + 128, 0, 1);
        }

        acc0 = __builtin_amdgcn_wmma_f32_16x16x32_bf16(
            false, a1.v, false, p1.v, (short)0, acc0, false, false);
        acc1 = __builtin_amdgcn_wmma_f32_16x16x32_bf16(
            false, a1.v, false, r1.v, (short)0, acc1, false, false);
    }

    // D layout: lane = N (mod 16); vgpr v -> M = v + 8*(lane>=16)
    int n  = n0 + (lane & 15);
    float bv0 = bias ? bias[n]      : 0.0f;
    float bv1 = bias ? bias[n + 16] : 0.0f;
    int mbase = m0 + ((lane < 16) ? 0 : 8);
    #pragma unroll
    for (int v = 0; v < 8; ++v) {
        float v0 = acc0[v] + bv0;
        float v1 = acc1[v] + bv1;
        if (act == 1) { v0 = fmaxf(v0, 0.0f); v1 = fmaxf(v1, 0.0f); }
        size_t o = (size_t)(mbase + v) * N + n;
        if (Cf) { Cf[o] = v0; Cf[o + 16] = v1; }
        if (Cb) { Cb[o] = (__bf16)v0; Cb[o + 16] = (__bf16)v1; }
    }
}

// ---------------- depthwise conv(4) + SiLU over xin = xz[:, :256] ----------------
__global__ void conv_silu_kernel(const __bf16* __restrict__ xz, const float* __restrict__ cw,
                                 const float* __restrict__ cb, __bf16* __restrict__ xc) {
    int idx = blockIdx.x * blockDim.x + threadIdx.x;   // NTOK*DINNER
    int d   = idx & (DINNER - 1);
    int tok = idx >> 8;
    int s   = tok & (SLEN - 1);
    float acc = cb[d];
    #pragma unroll
    for (int k = 0; k < DCONV; ++k) {
        int ss = s + k - (DCONV - 1);
        if (ss >= 0)
            acc += (float)xz[(size_t)(tok + k - (DCONV - 1)) * (2 * DINNER) + d] * cw[d * DCONV + k];
    }
    acc = acc * sigmoidf_(acc);
    xc[idx] = (__bf16)acc;
}

// ---------------- fused dt + selective scan + gate, last-token output ----------------
// grid = BSZ*4 blocks of 64 threads; block owns (b, 64-channel chunk).
// State (64 f32) and A-row live in VGPRs; per-step proj row shared via LDS.
__global__ void scan_kernel(const float* __restrict__ proj, const __bf16* __restrict__ xc,
                            const __bf16* __restrict__ xz, const float* __restrict__ dt_w,
                            const float* __restrict__ dt_b, const float* __restrict__ A_log,
                            const float* __restrict__ Dp, __bf16* __restrict__ ylast) {
    int t = threadIdx.x;                       // 0..63
    int b = blockIdx.x >> 2;
    int d = ((blockIdx.x & 3) << 6) + t;

    __shared__ float sAll[XPROJN];             // [0,8)=dt-rank, [8,72)=B, [72,136)=C

    float Arow[DSTATE], h[DSTATE], dtw[DTRANK];
    #pragma unroll
    for (int n = 0; n < DSTATE; ++n) { Arow[n] = -__expf(A_log[d * DSTATE + n]); h[n] = 0.0f; }
    #pragma unroll
    for (int r = 0; r < DTRANK; ++r) dtw[r] = dt_w[d * DTRANK + r];
    float dtb = dt_b[d];
    float Dd  = Dp[d];

    for (int s = 0; s < SLEN; ++s) {
        int tok = b * SLEN + s;
        for (int i = t; i < XPROJN; i += 64) sAll[i] = proj[(size_t)tok * XPROJP + i];
        __syncthreads();

        float dtraw = dtb;
        #pragma unroll
        for (int r = 0; r < DTRANK; ++r) dtraw += sAll[r] * dtw[r];
        float dt = (dtraw > 15.0f) ? dtraw : __logf(1.0f + __expf(dtraw));  // softplus
        float u  = (float)xc[(size_t)tok * DINNER + d];
        float du = dt * u;
        float y  = 0.0f;
        #pragma unroll
        for (int n = 0; n < DSTATE; ++n) {
            float dA = __expf(dt * Arow[n]);
            h[n] = dA * h[n] + du * sAll[DTRANK + n];
            y   += h[n] * sAll[DTRANK + DSTATE + n];
        }
        if (s == SLEN - 1) {
            float z   = (float)xz[(size_t)tok * (2 * DINNER) + DINNER + d];
            float out = (y + u * Dd) * (z * sigmoidf_(z));
            ylast[b * DINNER + d] = (__bf16)out;
        }
        __syncthreads();
    }
}

// ---------------- q = latent @ head_w^T + head_b (64x18, tiny) ----------------
__global__ void head_kernel(const float* __restrict__ latent, const float* __restrict__ hw,
                            const float* __restrict__ hb, float* __restrict__ q) {
    int idx = blockIdx.x * blockDim.x + threadIdx.x;
    if (idx >= BSZ * NACT) return;
    int b = idx / NACT, n = idx % NACT;
    float acc = hb[n];
    #pragma unroll 4
    for (int k = 0; k < DMODEL; ++k) acc += latent[b * DMODEL + k] * hw[n * DMODEL + k];
    q[idx] = acc;
}

extern "C" void kernel_launch(void* const* d_in, const int* in_sizes, int n_in,
                              void* d_out, int out_size, void* d_ws, size_t ws_size,
                              hipStream_t stream) {
    (void)in_sizes; (void)n_in; (void)out_size; (void)ws_size;
    const float* x        = (const float*)d_in[0];
    const float* W1       = (const float*)d_in[1];
    const float* b1       = (const float*)d_in[2];
    const float* W2       = (const float*)d_in[3];
    const float* b2       = (const float*)d_in[4];
    const float* in_proj  = (const float*)d_in[5];
    const float* conv_w   = (const float*)d_in[6];
    const float* conv_b   = (const float*)d_in[7];
    const float* x_proj   = (const float*)d_in[8];
    const float* dt_w     = (const float*)d_in[9];
    const float* dt_b     = (const float*)d_in[10];
    const float* A_log    = (const float*)d_in[11];
    const float* Dp       = (const float*)d_in[12];
    const float* out_proj = (const float*)d_in[13];
    const float* head_w   = (const float*)d_in[14];
    const float* head_b   = (const float*)d_in[15];

    float* q      = (float*)d_out;                 // [64,18]
    float* latent = (float*)d_out + BSZ * NACT;    // [64,128]

    // workspace carve-out (~105 MB)
    char* base = (char*)d_ws; size_t off = 0;
    auto alloc = [&](size_t bytes) -> void* {
        void* r = base + off; off += (bytes + 255) & ~(size_t)255; return r;
    };
    __bf16* xbf     = (__bf16*)alloc((size_t)NTOK * DIN * 2);
    __bf16* W1bf    = (__bf16*)alloc((size_t)MLPH * DIN * 2);
    __bf16* W2bf    = (__bf16*)alloc((size_t)DMODEL * MLPH * 2);
    __bf16* ipbf    = (__bf16*)alloc((size_t)2 * DINNER * DMODEL * 2);
    __bf16* xpbf    = (__bf16*)alloc((size_t)XPROJP * DINNER * 2);   // padded rows
    __bf16* opbf    = (__bf16*)alloc((size_t)DMODEL * DINNER * 2);
    __bf16* h1bf    = (__bf16*)alloc((size_t)NTOK * MLPH * 2);
    __bf16* h2bf    = (__bf16*)alloc((size_t)NTOK * DMODEL * 2);
    __bf16* xzbf    = (__bf16*)alloc((size_t)NTOK * 2 * DINNER * 2);
    __bf16* xcbf    = (__bf16*)alloc((size_t)NTOK * DINNER * 2);
    float*  projf   = (float*)alloc((size_t)NTOK * XPROJP * 4);      // padded cols
    __bf16* ylastbf = (__bf16*)alloc((size_t)BSZ * DINNER * 2);

    auto cvt = [&](const float* src, __bf16* dst, int n) {
        cvt_f32_to_bf16<<<(n + 255) / 256, 256, 0, stream>>>(src, dst, n);
    };
    cvt(x,        xbf,  NTOK * DIN);
    cvt(W1,       W1bf, MLPH * DIN);
    cvt(W2,       W2bf, DMODEL * MLPH);
    cvt(in_proj,  ipbf, 2 * DINNER * DMODEL);
    cvt(out_proj, opbf, DMODEL * DINNER);
    cvt_pad_bf16<<<(XPROJP * DINNER + 255) / 256, 256, 0, stream>>>(
        x_proj, xpbf, XPROJN, DINNER, XPROJP);

    auto gemm = [&](const __bf16* A, const __bf16* W, const float* bias,
                    float* Cf, __bf16* Cb, int M, int N, int K, int act) {
        int Mt = M / 16, Nt2 = N / 32;
        int tiles = Mt * Nt2;
        wmma_gemm<<<(tiles + 3) / 4, 128, 0, stream>>>(A, W, bias, Cf, Cb, M, N, K, Nt2, act);
    };

    // MLP1 -> MLP2 -> in_proj (all WMMA bf16, f32 accum)
    gemm(xbf,  W1bf, b1,      nullptr, h1bf, NTOK, MLPH,       DIN,    1);
    gemm(h1bf, W2bf, b2,      nullptr, h2bf, NTOK, DMODEL,     MLPH,   1);
    gemm(h2bf, ipbf, nullptr, nullptr, xzbf, NTOK, 2 * DINNER, DMODEL, 0);
    // depthwise conv + SiLU
    conv_silu_kernel<<<(NTOK * DINNER) / 256, 256, 0, stream>>>(xzbf, conv_w, conv_b, xcbf);
    // x_proj (padded to 160 output cols; scan reads with stride 160)
    gemm(xcbf, xpbf, nullptr, projf, nullptr, NTOK, XPROJP, DINNER, 0);
    // fused dt + scan + gating; emits only the last-token activations
    scan_kernel<<<BSZ * 4, 64, 0, stream>>>(projf, xcbf, xzbf, dt_w, dt_b, A_log, Dp, ylastbf);
    // out_proj on last token only -> latent, then head -> q
    gemm(ylastbf, opbf, nullptr, latent, nullptr, BSZ, DMODEL, DINNER, 0);
    head_kernel<<<(BSZ * NACT + 255) / 256, 256, 0, stream>>>(latent, head_w, head_b, q);
}
